// EdgeModel_72215580115031
// MI455X (gfx1250) — compile-verified, matrix operand order checked by reading, not run on hardware
//
#include <hip/hip_runtime.h>
#include <hip/hip_bf16.h>

// ---------------------------------------------------------------------------
// CDNA5 (gfx1250) fused MLP:  Y = LN(ReLU( LN(ReLU(X@W1+b1)) @ W2 + b2 ))
//  - bf16 WMMA (v_wmma_f32_16x16x32_bf16), fp32 accumulate
//  - weights prepacked to bf16 in fragment layout; B fragments loaded
//    DIRECTLY from global (L2-resident): no W staging, no phase-4 barriers
//  - B fragment loads issued BEFORE the staging barrier (latency hidden)
//  - MT=64 rows/workgroup: each B fragment feeds 4 WMMAs (A-reuse)
//  - sX double-buffered: one barrier per K-chunk in phase 1
//  - templated on (GATHER, DOUT): zero runtime branching in hot loops
//  - both LayerNorms done entirely in LDS (no hidden HBM round-trip)
// One workgroup = 8 wave32 waves, owns 64 rows x full N.
// ---------------------------------------------------------------------------

typedef __attribute__((ext_vector_type(16))) __bf16          v16bf;
typedef __attribute__((ext_vector_type(8)))  float           v8f;
typedef __attribute__((ext_vector_type(4)))  float           v4f;
typedef __attribute__((ext_vector_type(2)))  float           v2f;
typedef __attribute__((ext_vector_type(2)))  unsigned int    v2u;
typedef __attribute__((ext_vector_type(8)))  unsigned short  v8us;
typedef __attribute__((ext_vector_type(16))) unsigned short  v16us;

#define DHID   512          // hidden dim (fixed by the model)
#define MT     64           // rows per workgroup
#define KC     32           // K chunk per WMMA (bf16 K=32)
#define SXH    40           // LDS X row stride in halves (80B, 16B aligned)
#define SHH    (DHID + 8)   // LDS hidden row stride in halves (1040B)
#define SYF    264          // LDS Y row stride in floats (1056B)
// region offsets in ushort units (all 16B aligned)
#define OFF_X  0                       // 2 x 64*40 = 5120 (double buffered)
#define OFF_H  5120                    // 64*520    = 33280
#define OFF_Y  38400                   // 64*264 f32 = 33792 us
#define OFF_G  72192                   // 2304 f32 (g1,be1,b1|g2,be2,b2) = 4608 us
#define OFF_I  76800                   // 128 int = 256 us (512 us reserved) (gather indices)
#define SMEM_US   77312
#define SMEM_BYTES (SMEM_US * 2)       // 154624 B dynamic LDS (2 WGs/WGP fit)

__device__ __forceinline__ unsigned short f2bf(float f) {
    unsigned int u = __float_as_uint(f);
    u += 0x7FFFu + ((u >> 16) & 1u);      // round-to-nearest-even
    return (unsigned short)(u >> 16);
}
__device__ __forceinline__ float bf2f(unsigned short h) {
    return __uint_as_float(((unsigned int)h) << 16);
}

// Fragment = 16 bf16 per lane: two contiguous 16B reads at off, off+16
// (CDNA5 16-bit A/B VGPR layout, K-halves split by lane group).
// Works for LDS pointers (ds_load_b128) and global pointers (global_load_b128).
__device__ __forceinline__ v16bf ld_frag(const unsigned short* p, int off) {
    v8us lo = *(const v8us*)(p + off);
    v8us hi = *(const v8us*)(p + off + 16);
    v16us w;
#pragma unroll
    for (int i = 0; i < 8; ++i) { w[i] = lo[i]; w[8 + i] = hi[i]; }
    return __builtin_bit_cast(v16bf, w);
}

#define WMMA_BF16(A, B, C) \
    __builtin_amdgcn_wmma_f32_16x16x32_bf16(false, (A), false, (B), (short)0, (C), false, false)

// ---------------------------------------------------------------------------
// Weight prepack: P[((c*N + n)*32) + k] = bf16(W[(c*32 + k)*N + n])
// i.e. per 32-wide K-chunk c, W^T fragment rows [n][k] laid out contiguously.
// ---------------------------------------------------------------------------
extern "C" __global__ __launch_bounds__(256)
void prepack_w(const float* __restrict__ W, unsigned short* __restrict__ P,
               int K, int N, int lgN) {
    long long t = (long long)blockIdx.x * 256 + threadIdx.x;
    long long tot = (long long)K * N;
    if (t < tot) {
        int k = (int)(t & 31);
        long long cn = t >> 5;
        int n = (int)(cn & (N - 1));            // N is a power of two
        int c = (int)(cn >> lgN);
        P[t] = f2bf(W[(long long)(c * 32 + k) * N + n]);
    }
}

// ---------------------------------------------------------------------------
// Fused MLP kernel (GATHER: 0 = dense X, 1 = [srcA[iA]|srcB[iB]|srcE] concat)
// ---------------------------------------------------------------------------
template <int GATHER, int DOUT>
__global__ __launch_bounds__(256)
void mlp_fused_wmma(int M, int K1,
                    const float* __restrict__ X,
                    const float* __restrict__ srcA, const float* __restrict__ srcB,
                    const float* __restrict__ srcE,
                    const int* __restrict__ idxA, const int* __restrict__ idxB,
                    const unsigned short* __restrict__ PW1,
                    const float* __restrict__ b1,
                    const float* __restrict__ g1, const float* __restrict__ be1,
                    const unsigned short* __restrict__ PW2,
                    const float* __restrict__ b2,
                    const float* __restrict__ g2, const float* __restrict__ be2,
                    float* __restrict__ out)
{
    extern __shared__ unsigned short smem[];
    unsigned short* sX0 = smem + OFF_X;           // [2][64][SXH] bf16 X chunks
    unsigned short* sH  = smem + OFF_H;           // [64][SHH]    bf16 hidden
    float*          sY  = (float*)(smem + OFF_Y); // [64][SYF]    fp32 out tile
    float*          sGB = (float*)(smem + OFF_G); // g1|be1|b1|g2|be2|b2
    int*            sIx = (int*)(smem + OFF_I);   // 64 src + 64 dst indices

    const int tid  = threadIdx.x;
    const int lane = tid & 31;
    const int wave = tid >> 5;
    const int m0   = blockIdx.x * MT;
    const int lo   = (lane >> 4) << 3;            // K-half base per lane group
    const int ml   = lane & 15;

    // ---- prolog: stage norm params / biases / gather indices into LDS ----
    for (int i = tid; i < DHID; i += 256) {
        sGB[i]            = g1[i];
        sGB[DHID + i]     = be1[i];
        sGB[2 * DHID + i] = b1[i];
    }
    for (int i = tid; i < DOUT; i += 256) {
        sGB[3 * DHID + i]            = g2[i];
        sGB[3 * DHID + DOUT + i]     = be2[i];
        sGB[3 * DHID + 2 * DOUT + i] = b2[i];
    }
    if (GATHER && tid < 128) {
        int r = m0 + (tid & 63);
        r = (r < M) ? r : (M - 1);
        sIx[tid] = (tid < 64) ? idxA[r] : idxB[r];
    }

    const v8f vzero = {0.f, 0.f, 0.f, 0.f, 0.f, 0.f, 0.f, 0.f};
    v8f acc[16];
#pragma unroll
    for (int j = 0; j < 16; ++j) acc[j] = vzero;

    // helpers ------------------------------------------------------------
    auto loadB1 = [&](int kb, v16bf bfr[4]) {     // issued BEFORE the barrier
        const unsigned short* pw1c = PW1 + (long long)(kb >> 5) * (DHID * 32);
#pragma unroll
        for (int i = 0; i < 4; ++i)
            bfr[i] = ld_frag(pw1c + ((wave + (i << 3)) * 16 + ml) * 32, lo);
    };
    auto stage = [&](unsigned short* sX, auto addrfn) {
#pragma unroll
        for (int i = 0; i < 2; ++i) {
            int q  = tid + (i << 8);               // 0..511
            int r  = q >> 3;
            int k4 = (q & 7) << 2;
            int rgc = m0 + r; rgc = (rgc < M) ? rgc : (M - 1);
            v4f x = *(const v4f*)addrfn(r, rgc, k4);
            v2u pk;
            pk[0] = (unsigned)f2bf(x[0]) | ((unsigned)f2bf(x[1]) << 16);
            pk[1] = (unsigned)f2bf(x[2]) | ((unsigned)f2bf(x[3]) << 16);
            *(v2u*)&sX[r * SXH + k4] = pk;
        }
    };
    auto mmall = [&](unsigned short* sX, v16bf bfr[4]) {
#pragma unroll
        for (int m = 0; m < 4; ++m) {
            v16bf a = ld_frag(sX, (m * 16 + ml) * SXH + lo);
#pragma unroll
            for (int i = 0; i < 4; ++i)
                acc[m * 4 + i] = WMMA_BF16(a, bfr[i], acc[m * 4 + i]);
        }
    };

    // ---------------- Phase 1: H_acc = X @ W1 (K1 -> 512) ----------------
    int parity = 0;
    if constexpr (!GATHER) {
        for (int kb = 0; kb < K1; kb += KC, parity ^= 1) {
            unsigned short* sX = sX0 + parity * (MT * SXH);
            v16bf bfr[4];
            loadB1(kb, bfr);
            stage(sX, [&](int, int rgc, int k4)
                      { return X + (long long)rgc * K1 + (kb + k4); });
            __syncthreads();
            mmall(sX, bfr);
        }
    } else {
        // K1 == 768: three statically-selected 256-wide concat segments
        for (int kb = 0; kb < 256; kb += KC, parity ^= 1) {
            unsigned short* sX = sX0 + parity * (MT * SXH);
            v16bf bfr[4];
            loadB1(kb, bfr);
            stage(sX, [&](int r, int, int k4)
                      { return srcA + (long long)sIx[r] * 256 + (kb + k4); });
            __syncthreads();
            mmall(sX, bfr);
        }
        for (int kb = 256; kb < 512; kb += KC, parity ^= 1) {
            unsigned short* sX = sX0 + parity * (MT * SXH);
            v16bf bfr[4];
            loadB1(kb, bfr);
            stage(sX, [&](int r, int, int k4)
                      { return srcB + (long long)sIx[64 + r] * 256 + (kb - 256 + k4); });
            __syncthreads();
            mmall(sX, bfr);
        }
        for (int kb = 512; kb < 768; kb += KC, parity ^= 1) {
            unsigned short* sX = sX0 + parity * (MT * SXH);
            v16bf bfr[4];
            loadB1(kb, bfr);
            stage(sX, [&](int, int rgc, int k4)
                      { return srcE + (long long)rgc * 256 + (kb - 512 + k4); });
            __syncthreads();
            mmall(sX, bfr);
        }
    }
    __syncthreads();

    // ---------------- Phase 2: bias + ReLU -> sH (bf16) ------------------
    {
        int mr = (lane >> 4) << 3;
#pragma unroll
        for (int m = 0; m < 4; ++m)
#pragma unroll
            for (int i = 0; i < 4; ++i) {
                int nb = wave + (i << 3);
#pragma unroll
                for (int v = 0; v < 8; ++v) {
                    int row = m * 16 + mr + v;
                    int col = nb * 16 + ml;
                    float x = acc[m * 4 + i][v] + sGB[2 * DHID + col];
                    sH[row * SHH + col] = f2bf(x > 0.f ? x : 0.f);
                }
            }
    }
    __syncthreads();

    // ---------------- Phase 3: LayerNorm over 512 (in LDS) ---------------
#pragma unroll
    for (int r = 0; r < 8; ++r) {
        int row = wave * 8 + r;
        float s = 0.f, s2 = 0.f;
        unsigned pr[8];
#pragma unroll
        for (int u = 0; u < 8; ++u) {              // paired b32 loads
            pr[u] = *(const unsigned*)&sH[row * SHH + ((lane + (u << 5)) << 1)];
            float v0 = bf2f((unsigned short)pr[u]);
            float v1 = bf2f((unsigned short)(pr[u] >> 16));
            s += v0 + v1; s2 += v0 * v0 + v1 * v1;
        }
#pragma unroll
        for (int off = 16; off > 0; off >>= 1) {
            s  += __shfl_xor(s,  off, 32);
            s2 += __shfl_xor(s2, off, 32);
        }
        float mean = s * (1.f / DHID);
        float var  = s2 * (1.f / DHID) - mean * mean;
        float rstd = rsqrtf(var + 1e-5f);
#pragma unroll
        for (int u = 0; u < 8; ++u) {
            int c0 = (lane + (u << 5)) << 1;
            float v0 = (bf2f((unsigned short)pr[u])         - mean) * rstd * sGB[c0]     + sGB[DHID + c0];
            float v1 = (bf2f((unsigned short)(pr[u] >> 16)) - mean) * rstd * sGB[c0 + 1] + sGB[DHID + c0 + 1];
            *(unsigned*)&sH[row * SHH + c0] =
                (unsigned)f2bf(v0) | ((unsigned)f2bf(v1) << 16);
        }
    }
    __syncthreads();

    // ------- Phase 4: Y_acc = H @ W2 (512 -> DOUT), barrier-free ---------
    constexpr int nb2   = DOUT / 16;      // N-blocks (16 or 4)
    constexpr int lb2   = (DOUT == 256) ? 4 : 2;
    constexpr int j2max = nb2 / 2;        // tiles per wave (8 or 2)
#pragma unroll
    for (int j = 0; j < j2max; ++j) acc[j] = vzero;

    for (int kb = 0; kb < DHID; kb += KC) {
        const unsigned short* pw2c = PW2 + (long long)(kb >> 5) * (DOUT * 32);
        // the wave touches at most 2 distinct B tiles per chunk
        v16bf bA = ld_frag(pw2c + (((wave)     & (nb2 - 1)) * 16 + ml) * 32, lo);
        v16bf bB = bA;
        if constexpr (j2max > 2)
            bB = ld_frag(pw2c + (((wave + 8) & (nb2 - 1)) * 16 + ml) * 32, lo);
#pragma unroll
        for (int j2 = 0; j2 < j2max; ++j2) {
            int tt = wave + (j2 << 3);
            int mb = tt >> lb2;
            v16bf a = ld_frag(sH, (mb * 16 + ml) * SHH + kb + lo);
            acc[j2] = WMMA_BF16(a, (j2 & 1) ? bB : bA, acc[j2]);
        }
    }

    // ---------------- Phase 5: bias + ReLU -> sY, LN, store --------------
    {
        int mr = (lane >> 4) << 3;
#pragma unroll
        for (int j2 = 0; j2 < j2max; ++j2) {
            int tt = wave + (j2 << 3);
            int mb = tt >> lb2, nb = tt & (nb2 - 1);
#pragma unroll
            for (int v = 0; v < 8; ++v) {
                int row = mb * 16 + mr + v;
                int col = nb * 16 + ml;
                float x = acc[j2][v] + sGB[3 * DHID + 2 * DOUT + col];
                sY[row * SYF + col] = x > 0.f ? x : 0.f;
            }
        }
    }
    __syncthreads();

    constexpr float rdout = 1.f / (float)DOUT;
    constexpr int   upr   = DOUT / 64;    // float2 pairs per lane per row
#pragma unroll
    for (int r = 0; r < 8; ++r) {
        int row = wave * 8 + r;
        int rg  = m0 + row;
        float s = 0.f, s2 = 0.f;
#pragma unroll
        for (int u = 0; u < upr; ++u) {
            v2f v = *(const v2f*)&sY[row * SYF + ((lane + (u << 5)) << 1)];
            s += v[0] + v[1]; s2 += v[0] * v[0] + v[1] * v[1];
        }
#pragma unroll
        for (int off = 16; off > 0; off >>= 1) {
            s  += __shfl_xor(s,  off, 32);
            s2 += __shfl_xor(s2, off, 32);
        }
        float mean = s * rdout;
        float var  = s2 * rdout - mean * mean;
        float rstd = rsqrtf(var + 1e-5f);
        if (rg < M) {
#pragma unroll
            for (int u = 0; u < upr; ++u) {
                int c0 = (lane + (u << 5)) << 1;
                v2f v = *(const v2f*)&sY[row * SYF + c0];
                v2f o;
                o[0] = (v[0] - mean) * rstd * sGB[3 * DHID + c0]     + sGB[3 * DHID + DOUT + c0];
                o[1] = (v[1] - mean) * rstd * sGB[3 * DHID + c0 + 1] + sGB[3 * DHID + DOUT + c0 + 1];
                *(v2f*)&out[(long long)rg * DOUT + c0] = o;
            }
        }
    }
}

// ---------------------------------------------------------------------------
// Host side
// ---------------------------------------------------------------------------
struct P8 { const float *W1, *b1, *g1, *be1, *W2, *b2, *g2, *be2; };

static P8 p8_list(void* const* d, int base) {
    P8 p;
    p.W1  = (const float*)d[base + 0]; p.b1  = (const float*)d[base + 1];
    p.g1  = (const float*)d[base + 2]; p.be1 = (const float*)d[base + 3];
    p.W2  = (const float*)d[base + 4]; p.b2  = (const float*)d[base + 5];
    p.g2  = (const float*)d[base + 6]; p.be2 = (const float*)d[base + 7];
    return p;
}
static P8 p8_blob(const void* blob, long long din, long long dout) {
    const float* f = (const float*)blob;
    P8 p;
    p.W1 = f;  f += din * 512;  p.b1 = f;  f += 512;
    p.g1 = f;  f += 512;        p.be1 = f; f += 512;
    p.W2 = f;  f += 512 * dout; p.b2 = f;  f += dout;
    p.g2 = f;  f += dout;       p.be2 = f; f += dout;
    return p;
}
static int ilog2(int v) { int l = 0; while ((1 << l) < v) ++l; return l; }

extern "C" void kernel_launch(void* const* d_in, const int* in_sizes, int n_in,
                              void* d_out, int out_size, void* d_ws, size_t ws_size,
                              hipStream_t stream) {
    const float* node_ins = (const float*)d_in[0];
    const int*   ei_ins   = (const int*)  d_in[1];
    const float* ea_ins   = (const float*)d_in[2];
    const float* node_lab = (const float*)d_in[3];
    const int*   ei_lab   = (const int*)  d_in[4];
    const float* ea_lab   = (const float*)d_in[5];
    const int*   ei_crx   = (const int*)  d_in[6];
    const float* ea_crx   = (const float*)d_in[7];

    const long long Nins = in_sizes[0] / 128;   // node counts
    const long long Nlab = in_sizes[3] / 128;
    const long long Eins = in_sizes[2] / 64;    // edge counts
    const long long Elab = in_sizes[5] / 64;
    const long long Ecrx = in_sizes[7] / 64;

    const bool flat = (n_in >= 72);             // tuple params flattened?
    P8 pp[8];
    pp[0] = flat ? p8_list(d_in, 8)  : p8_blob(d_in[8],  128, 256);  // node enc ins
    pp[1] = flat ? p8_list(d_in, 16) : p8_blob(d_in[9],   64, 256);  // edge enc ins
    pp[2] = flat ? p8_list(d_in, 24) : p8_blob(d_in[10], 128, 256);  // node enc lab
    pp[3] = flat ? p8_list(d_in, 32) : p8_blob(d_in[11],  64, 256);  // edge enc lab
    pp[4] = flat ? p8_list(d_in, 40) : p8_blob(d_in[12],  64, 256);  // edge enc crx
    pp[5] = flat ? p8_list(d_in, 48) : p8_blob(d_in[13], 768,  64);  // edge mlp ins
    pp[6] = flat ? p8_list(d_in, 56) : p8_blob(d_in[14], 768,  64);  // edge mlp lab
    pp[7] = flat ? p8_list(d_in, 64) : p8_blob(d_in[15], 768,  64);  // edge mlp crx
    const int K1s[8] = {128, 64, 128, 64, 64, 768, 768, 768};
    const int DOs[8] = {256, 256, 256, 256, 256, 64, 64, 64};

    // workspace layout: prepacked bf16 weights, then encoded-edge buffer
    unsigned short* wbase = (unsigned short*)d_ws;
    long long woff = 0;
    unsigned short* PW1[8]; unsigned short* PW2[8];
    for (int m = 0; m < 8; ++m) {
        PW1[m] = wbase + woff; woff += (long long)K1s[m] * 512;
        PW2[m] = wbase + woff; woff += 512LL * DOs[m];
    }
    float* ebuf = (float*)(wbase + ((woff + 7) & ~7LL)); // 16B-aligned

    // prepack all weights (runs every call; ~4.3MB total, trivial cost)
    for (int m = 0; m < 8; ++m) {
        long long t1 = (long long)K1s[m] * 512;
        long long t2 = 512LL * DOs[m];
        prepack_w<<<dim3((unsigned)((t1 + 255) / 256)), dim3(256), 0, stream>>>(
            pp[m].W1, PW1[m], K1s[m], 512, 9);
        prepack_w<<<dim3((unsigned)((t2 + 255) / 256)), dim3(256), 0, stream>>>(
            pp[m].W2, PW2[m], 512, DOs[m], ilog2(DOs[m]));
    }

    float* outp = (float*)d_out;                 // outputs concatenated flat
    float* n_ins_o   = outp;
    float* out_ins_o = n_ins_o   + Nins * 256;
    float* n_lab_o   = out_ins_o + Eins * 64;
    float* out_lab_o = n_lab_o   + Nlab * 256;
    float* out_crx_o = out_lab_o + Elab * 64;

    auto Lenc = [&](long long M, int m, const float* X, float* o) {
        dim3 grid((unsigned)((M + MT - 1) / MT)), blk(256);
        mlp_fused_wmma<0, 256><<<grid, blk, SMEM_BYTES, stream>>>(
            (int)M, K1s[m], X, nullptr, nullptr, nullptr, nullptr, nullptr,
            PW1[m], pp[m].b1, pp[m].g1, pp[m].be1,
            PW2[m], pp[m].b2, pp[m].g2, pp[m].be2, o);
    };
    auto Ledge = [&](long long M, int m, const float* sA, const float* sB,
                     const float* sE, const int* iA, const int* iB, float* o) {
        dim3 grid((unsigned)((M + MT - 1) / MT)), blk(256);
        mlp_fused_wmma<1, 64><<<grid, blk, SMEM_BYTES, stream>>>(
            (int)M, 768, nullptr, sA, sB, sE, iA, iB,
            PW1[m], pp[m].b1, pp[m].g1, pp[m].be1,
            PW2[m], pp[m].b2, pp[m].g2, pp[m].be2, o);
    };

    // node encoders
    Lenc(Nins, 0, node_ins, n_ins_o);
    Lenc(Nlab, 2, node_lab, n_lab_o);

    // instance edges: encode then edge-conv (gather [n_ins[r]|n_ins[c]|e])
    Lenc(Eins, 1, ea_ins, ebuf);
    Ledge(Eins, 5, n_ins_o, n_ins_o, ebuf, ei_ins, ei_ins + Eins, out_ins_o);

    // label edges
    Lenc(Elab, 3, ea_lab, ebuf);
    Ledge(Elab, 6, n_lab_o, n_lab_o, ebuf, ei_lab, ei_lab + Elab, out_lab_o);

    // cross edges (src = ins nodes, dst = label nodes)
    Lenc(Ecrx, 4, ea_crx, ebuf);
    Ledge(Ecrx, 7, n_ins_o, n_lab_o, ebuf, ei_crx, ei_crx + Ecrx, out_crx_o);

    (void)out_size; (void)ws_size;
}
